// GoldenMoEDynamicI_9981503995949
// MI455X (gfx1250) — compile-verified, matrix operand order checked by reading, not run on hardware
//
#include <hip/hip_runtime.h>
#include <hip/hip_bf16.h>
#include <math.h>

typedef __attribute__((ext_vector_type(16))) __bf16 v16bf;
typedef __attribute__((ext_vector_type(8)))  __bf16 v8bf;
typedef __attribute__((ext_vector_type(2)))  __bf16 v2bf;
typedef __attribute__((ext_vector_type(2)))  float  v2f;
typedef __attribute__((ext_vector_type(8)))  float  v8f;

#define BATCH 4096
#define DIN   1024
#define DHID  4096
#define DOUTD 1024
#define NEXP  8
#define TEMP  2.718281828459045f

// ---- optional CDNA5 async global->LDS path (guarded; falls back cleanly) ----
#if defined(__has_builtin)
#if __has_builtin(__builtin_amdgcn_global_load_async_to_lds_b128)
#define HAVE_ASYNC_B128 1
#endif
#if __has_builtin(__builtin_amdgcn_s_wait_asynccnt)
#define HAVE_WAIT_ASYNC 1
#endif
#endif

#ifdef HAVE_ASYNC_B128
typedef int gint4 __attribute__((vector_size(16)));
typedef __attribute__((address_space(1))) gint4* as1_i4p;
typedef __attribute__((address_space(3))) gint4* as3_i4p;
#define ASYNC_CP_B128(gp, lp) \
  __builtin_amdgcn_global_load_async_to_lds_b128((as1_i4p)(gp), (as3_i4p)(lp), 0, 0)
#endif

__device__ __forceinline__ void async_wait0() {
#ifdef HAVE_ASYNC_B128
#ifdef HAVE_WAIT_ASYNC
  __builtin_amdgcn_s_wait_asynccnt(0);
#else
  asm volatile("s_wait_asynccnt 0x0" ::: "memory");
#endif
#endif
}

// ---------------- helpers (native HW bf16 conversion) ----------------

__device__ __forceinline__ unsigned short f2bf_bits(float f) {
  union { __bf16 b; unsigned short s; } u;
  u.b = (__bf16)f;                 // lowers to v_cvt_pk_bf16_f32 (RNE)
  return u.s;
}

// packed f32x2 -> bf16x2: vector fptrunc -> single v_cvt_pk_bf16_f32
__device__ __forceinline__ unsigned pack2bf(float a, float b) {
  v2f f; f.x = a; f.y = b;
  union { v2bf v; unsigned u; } c;
  c.v = __builtin_convertvector(f, v2bf);
  return c.u;
}

// 16x32 bf16 fragment load (CDNA5 ISA 7.12.2 layout): lanes 0-15 hold row
// M=lane, K=0..7 (v0-3) + 16..23 (v4-7); lanes 16-31 hold K=8..15 / 24..31.
__device__ __forceinline__ v16bf ldsFrag(const unsigned short* base, int rowBase,
                                         int lane) {
  int r  = rowBase + (lane & 15);
  int kb = (lane >> 4) << 3;
  const v8bf* p0 = (const v8bf*)(base + r * 40 + kb);
  const v8bf* p1 = (const v8bf*)(base + r * 40 + 16 + kb);
  v8bf lo = *p0;
  v8bf hi = *p1;
  return __builtin_shufflevector(lo, hi, 0,1,2,3,4,5,6,7,8,9,10,11,12,13,14,15);
}

// ---------------- small kernels ----------------

__global__ void moe_init(float* nact) {
  if (threadIdx.x == 0 && blockIdx.x == 0) *nact = 0.0f;
}

// f32 -> bf16 bulk convert (8 elems/thread)
__global__ __launch_bounds__(256) void cvt_bf16(
    const float* __restrict__ src, unsigned short* __restrict__ dst) {
  int i = (blockIdx.x * 256 + threadIdx.x) * 8;
  float4 a = *(const float4*)(src + i);
  float4 b = *(const float4*)(src + i + 4);
  uint4 q;
  q.x = pack2bf(a.x, a.y); q.y = pack2bf(a.z, a.w);
  q.z = pack2bf(b.x, b.y); q.w = pack2bf(b.z, b.w);
  *(uint4*)(dst + i) = q;
}

// one wave per token: gate softmax(T=e) + clarity MLP -> n_active atomicAdd
__global__ __launch_bounds__(256) void moe_gate(
    const float* __restrict__ X,  const float* __restrict__ GW,
    const float* __restrict__ GB, const float* __restrict__ CW1,
    const float* __restrict__ CB1,const float* __restrict__ CW2,
    const float* __restrict__ CB2,float* __restrict__ probs,
    float* __restrict__ nact_sum)
{
  const int wave  = threadIdx.x >> 5;
  const int lane  = threadIdx.x & 31;
  const int token = blockIdx.x * 8 + wave;
  const float* x  = X + (size_t)token * DIN;

  float g[NEXP];  float c[16];
  #pragma unroll
  for (int e = 0; e < NEXP; e++) g[e] = 0.0f;
  #pragma unroll
  for (int j = 0; j < 16; j++)   c[j] = 0.0f;

  for (int d = lane; d < DIN; d += 32) {
    float xv = x[d];
    const float* gw = GW  + (size_t)d * NEXP;
    #pragma unroll
    for (int e = 0; e < NEXP; e++) g[e] += xv * gw[e];
    const float* cw = CW1 + (size_t)d * 16;
    #pragma unroll
    for (int j = 0; j < 16; j++)   c[j] += xv * cw[j];
  }
  #pragma unroll
  for (int off = 16; off > 0; off >>= 1) {
    #pragma unroll
    for (int e = 0; e < NEXP; e++) g[e] += __shfl_xor(g[e], off);
    #pragma unroll
    for (int j = 0; j < 16; j++)   c[j] += __shfl_xor(c[j], off);
  }

  if (lane == 0) {
    float s[NEXP];
    float mx = -1e30f;
    #pragma unroll
    for (int e = 0; e < NEXP; e++) { s[e] = (g[e] + GB[e]) / TEMP; mx = fmaxf(mx, s[e]); }
    float tot = 0.0f;
    #pragma unroll
    for (int e = 0; e < NEXP; e++) { s[e] = expf(s[e] - mx); tot += s[e]; }
    float inv = 1.0f / tot;
    #pragma unroll
    for (int e = 0; e < NEXP; e++) probs[(size_t)token * NEXP + e] = s[e] * inv;

    float z = CB2[0];
    #pragma unroll
    for (int j = 0; j < 16; j++) {
      float h = c[j] + CB1[j];
      h = h > 0.0f ? h : 0.0f;
      z += h * CW2[j];
    }
    float sig = 1.0f / (1.0f + expf(-z));
    atomicAdd(nact_sum, (float)NEXP - sig * (float)(NEXP - 2));
  }
}

// one thread per token: k from batch mean, stable top-k mask, renormalize
__global__ __launch_bounds__(256) void moe_weights(
    const float* __restrict__ probs, const float* __restrict__ nact_sum,
    float* __restrict__ WT)
{
  int token = blockIdx.x * blockDim.x + threadIdx.x;
  if (token >= BATCH) return;
  float mean = *nact_sum / (float)BATCH;
  int k = (int)floorf(mean + 0.5f);
  k = k < 2 ? 2 : (k > NEXP ? NEXP : k);

  float p[NEXP];
  #pragma unroll
  for (int i = 0; i < NEXP; i++) p[i] = probs[(size_t)token * NEXP + i];

  float w[NEXP]; float tot = 0.0f;
  #pragma unroll
  for (int i = 0; i < NEXP; i++) {
    int rank = 0;
    #pragma unroll
    for (int j = 0; j < NEXP; j++)
      rank += (p[j] > p[i]) || (p[j] == p[i] && j < i);
    w[i] = (rank < k) ? p[i] : 0.0f;
    tot += w[i];
  }
  float inv = 1.0f / (tot + 1e-8f);
  #pragma unroll
  for (int i = 0; i < NEXP; i++) WT[(size_t)token * NEXP + i] = w[i] * inv;
}

// ---------------- fused expert GEMM (bf16 WMMA, f32 accumulate) ----------------
// Block tile 256x128, BK=32, 8 waves, each wave 64x64 (4x4 WMMA frags).
// EPI 0: outH = relu(A@W + bias) as bf16.  EPI 1: outF (+)= w[:,e]*(A@W + bias).
template <int EPI>
__global__ __launch_bounds__(256) void moe_gemm(
    const unsigned short* __restrict__ A,   // bf16 [M,K]
    const float* __restrict__ W,            // f32  [K,N]
    const float* __restrict__ bias,         // [N]
    const float* __restrict__ WT,           // [BATCH,NEXP] (EPI 1)
    float* __restrict__ outF,               // [M,N] (EPI 1)
    unsigned short* __restrict__ outH,      // bf16 [M,N] (EPI 0)
    int K, int N, int expert, int accumulate)
{
  __shared__ alignas(16) unsigned short sA[2][256 * 40];
  __shared__ alignas(16) unsigned short sB[2][128 * 40];  // transposed [n][k]

  const int t = threadIdx.x;
  const int lane = t & 31, wave = t >> 5;
  const int waveM = wave >> 1, waveN = wave & 1;
  const int m0 = blockIdx.y * 256, n0 = blockIdx.x * 128;

  v8f acc[4][4];
  #pragma unroll
  for (int m = 0; m < 4; m++)
    #pragma unroll
    for (int n = 0; n < 4; n++)
      #pragma unroll
      for (int q = 0; q < 8; q++) acc[m][n][q] = 0.0f;

  // staging roles
  const unsigned short* gA0 = A + (size_t)(m0 + t) * K;        // 1 row/thread
  const int bKg = (t >> 5) << 2;                                // k0: 0,4,..28
  const int bNg = (t & 31) << 2;                                // n:  0,4,..124

  const int NT = K / 32;

  // ---- stage tile 0 ----
  {
    unsigned short* lA = &sA[0][t * 40];
#ifdef HAVE_ASYNC_B128
    #pragma unroll
    for (int j = 0; j < 4; j++)
      ASYNC_CP_B128(gA0 + j * 8, lA + j * 8);
#else
    #pragma unroll
    for (int j = 0; j < 4; j++)
      ((uint4*)lA)[j] = ((const uint4*)gA0)[j];
#endif
    const float* gB = W + (size_t)bKg * N + n0 + bNg;
    float4 f0 = *(const float4*)(gB);
    float4 f1 = *(const float4*)(gB + N);
    float4 f2 = *(const float4*)(gB + 2 * N);
    float4 f3 = *(const float4*)(gB + 3 * N);
    float r0[4] = {f0.x, f0.y, f0.z, f0.w};
    float r1[4] = {f1.x, f1.y, f1.z, f1.w};
    float r2[4] = {f2.x, f2.y, f2.z, f2.w};
    float r3[4] = {f3.x, f3.y, f3.z, f3.w};
    #pragma unroll
    for (int j = 0; j < 4; j++) {
      uint2 v; v.x = pack2bf(r0[j], r1[j]); v.y = pack2bf(r2[j], r3[j]);
      *(uint2*)(&sB[0][(bNg + j) * 40 + bKg]) = v;
    }
  }
  async_wait0();
  __syncthreads();

  for (int it = 0; it < NT; ++it) {
    const int cur = it & 1, nxt = cur ^ 1;
    const bool more = (it + 1 < NT);
    const int ktn = (it + 1) * 32;

    float r0[4], r1[4], r2[4], r3[4];
#ifndef HAVE_ASYNC_B128
    uint4 areg[4];
#endif
    if (more) {
#ifdef HAVE_ASYNC_B128
      unsigned short* lA = &sA[nxt][t * 40];
      #pragma unroll
      for (int j = 0; j < 4; j++)
        ASYNC_CP_B128(gA0 + ktn + j * 8, lA + j * 8);
#else
      #pragma unroll
      for (int j = 0; j < 4; j++)
        areg[j] = ((const uint4*)(gA0 + ktn))[j];
#endif
      const float* gB = W + (size_t)(ktn + bKg) * N + n0 + bNg;
      float4 f0 = *(const float4*)(gB);
      float4 f1 = *(const float4*)(gB + N);
      float4 f2 = *(const float4*)(gB + 2 * N);
      float4 f3 = *(const float4*)(gB + 3 * N);
      r0[0]=f0.x; r0[1]=f0.y; r0[2]=f0.z; r0[3]=f0.w;
      r1[0]=f1.x; r1[1]=f1.y; r1[2]=f1.z; r1[3]=f1.w;
      r2[0]=f2.x; r2[1]=f2.y; r2[2]=f2.z; r2[3]=f2.w;
      r3[0]=f3.x; r3[1]=f3.y; r3[2]=f3.z; r3[3]=f3.w;
    }

    // ---- compute current tile ----
    v16bf af[4], bfr[4];
    #pragma unroll
    for (int m = 0; m < 4; m++) af[m]  = ldsFrag(&sA[cur][0], waveM * 64 + m * 16, lane);
    #pragma unroll
    for (int n = 0; n < 4; n++) bfr[n] = ldsFrag(&sB[cur][0], waveN * 64 + n * 16, lane);
    #pragma unroll
    for (int m = 0; m < 4; m++)
      #pragma unroll
      for (int n = 0; n < 4; n++)
        acc[m][n] = __builtin_amdgcn_wmma_f32_16x16x32_bf16(
            false, af[m], false, bfr[n], (short)0, acc[m][n], false, false);

    if (more) {
#ifndef HAVE_ASYNC_B128
      unsigned short* lA = &sA[nxt][t * 40];
      #pragma unroll
      for (int j = 0; j < 4; j++) ((uint4*)lA)[j] = areg[j];
#endif
      #pragma unroll
      for (int j = 0; j < 4; j++) {
        uint2 v; v.x = pack2bf(r0[j], r1[j]); v.y = pack2bf(r2[j], r3[j]);
        *(uint2*)(&sB[nxt][(bNg + j) * 40 + bKg]) = v;
      }
    }
    async_wait0();
    __syncthreads();
  }

  // ---- epilogue ----
  #pragma unroll
  for (int m = 0; m < 4; m++) {
    int row0 = m0 + waveM * 64 + m * 16 + ((lane >> 4) << 3);
    #pragma unroll
    for (int n = 0; n < 4; n++) {
      int col = n0 + waveN * 64 + n * 16 + (lane & 15);
      float bv = bias[col];
      #pragma unroll
      for (int i = 0; i < 8; i++) {
        int row = row0 + i;
        float v = acc[m][n][i] + bv;
        if (EPI == 0) {
          v = v > 0.0f ? v : 0.0f;
          outH[(size_t)row * N + col] = f2bf_bits(v);
        } else {
          float wgt = WT[(size_t)row * NEXP + expert];
          size_t idx = (size_t)row * N + col;
          float r = v * wgt;
          if (accumulate) outF[idx] += r;
          else            outF[idx]  = r;
        }
      }
    }
  }
}

// ---------------- launch ----------------

extern "C" void kernel_launch(void* const* d_in, const int* in_sizes, int n_in,
                              void* d_out, int out_size, void* d_ws, size_t ws_size,
                              hipStream_t stream) {
  (void)in_sizes; (void)n_in; (void)out_size; (void)ws_size;
  const float* x   = (const float*)d_in[0];
  const float* gw  = (const float*)d_in[1];
  const float* gb  = (const float*)d_in[2];
  const float* cw1 = (const float*)d_in[3];
  const float* cb1 = (const float*)d_in[4];
  const float* cw2 = (const float*)d_in[5];
  const float* cb2 = (const float*)d_in[6];
  const float* ew1 = (const float*)d_in[7];
  const float* eb1 = (const float*)d_in[8];
  const float* ew2 = (const float*)d_in[9];
  const float* eb2 = (const float*)d_in[10];
  float* out = (float*)d_out;

  float* ws    = (float*)d_ws;
  float* nact  = ws;                                   // 1 f32 (padded)
  float* probs = ws + 64;                              // BATCH*NEXP
  float* wts   = probs + BATCH * NEXP;                 // BATCH*NEXP
  unsigned short* H   = (unsigned short*)(wts + BATCH * NEXP);   // bf16 [BATCH,DHID]
  unsigned short* xbf = H + (size_t)BATCH * DHID;                // bf16 [BATCH,DIN]

  moe_init<<<dim3(1), dim3(1), 0, stream>>>(nact);
  cvt_bf16<<<dim3((BATCH * DIN) / (256 * 8)), dim3(256), 0, stream>>>(x, xbf);
  moe_gate<<<dim3(BATCH / 8), dim3(256), 0, stream>>>(
      x, gw, gb, cw1, cb1, cw2, cb2, probs, nact);
  moe_weights<<<dim3(BATCH / 256), dim3(256), 0, stream>>>(probs, nact, wts);

  for (int e = 0; e < NEXP; e++) {
    moe_gemm<0><<<dim3(DHID / 128, BATCH / 256), dim3(256), 0, stream>>>(
        xbf, ew1 + (size_t)e * DIN * DHID, eb1 + (size_t)e * DHID,
        nullptr, nullptr, H, DIN, DHID, e, 0);
    moe_gemm<1><<<dim3(DOUTD / 128, BATCH / 256), dim3(256), 0, stream>>>(
        H, ew2 + (size_t)e * DHID * DOUTD, eb2 + (size_t)e * DOUTD,
        wts, out, nullptr, DHID, DOUTD, e, e > 0 ? 1 : 0);
  }
}
